// SynthesisLayer_38878043963808
// MI455X (gfx1250) — matrix operand order repaired and checked by reading, not compiled, not run on hardware
//
#include <hip/hip_runtime.h>

// ---------------------------------------------------------------------------
// Problem constants
// ---------------------------------------------------------------------------
#define BATCH     8
#define CH        512            // C_IN == C_OUT == HIDDEN == W_DIM
#define HW        4096           // 64*64
#define NTAPS     28             // low_filter length
#define INV_SQRT512 0.044194173824159216f
#define SQRT2       1.4142135623730951f

typedef __attribute__((ext_vector_type(16))) __bf16 v16bf;
typedef __attribute__((ext_vector_type(8)))  float  v8f;

union FragBf { uint4 q[2]; v16bf v; };

static __device__ __forceinline__ unsigned short f2bf(float f) {
    unsigned u = __float_as_uint(f);
    unsigned r = u + 0x7FFFu + ((u >> 16) & 1u);   // round-to-nearest-even
    return (unsigned short)(r >> 16);
}
static __device__ __forceinline__ unsigned short to_bf(float v)          { return f2bf(v); }
static __device__ __forceinline__ unsigned short to_bf(unsigned short v) { return v; }

// CDNA5 async global->LDS copy (16B per lane), tracked by ASYNCcnt.
static __device__ __forceinline__ void async_copy_b128(unsigned lds_off, const void* gptr) {
    asm volatile("global_load_async_to_lds_b128 %0, %1, off"
                 :: "v"(lds_off), "v"((unsigned long long)(uintptr_t)gptr)
                 : "memory");
}
static __device__ __forceinline__ void wait_async0() {
    asm volatile("s_wait_asynccnt 0" ::: "memory");
}

// ---------------------------------------------------------------------------
// 1) s[b][c] = dot(w_latent[b], affine_w[c]) / sqrt(512) + affine_b[c]
// ---------------------------------------------------------------------------
__global__ __launch_bounds__(256) void k_affine(const float* __restrict__ wl,
                                                const float* __restrict__ aw,
                                                const float* __restrict__ ab,
                                                float* __restrict__ s) {
    int idx = blockIdx.x * 256 + threadIdx.x;      // 4096 = 8*512
    int b = idx >> 9, c = idx & 511;
    const float* wr = wl + (size_t)b * CH;
    const float* ar = aw + (size_t)c * CH;
    float acc = 0.f;
    for (int k = 0; k < CH; ++k) acc += wr[k] * ar[k];
    s[idx] = acc * INV_SQRT512 + ab[c];
}

// ---------------------------------------------------------------------------
// 2) sn_scale = rsqrt(mean(s^2))  (deterministic single-block tree reduce)
// ---------------------------------------------------------------------------
__global__ __launch_bounds__(256) void k_s_reduce(const float* __restrict__ s,
                                                  float* __restrict__ out) {
    __shared__ float red[256];
    float a = 0.f;
    for (int j = 0; j < 16; ++j) { float v = s[threadIdx.x + 256 * j]; a += v * v; }
    red[threadIdx.x] = a; __syncthreads();
    for (int st = 128; st > 0; st >>= 1) {
        if (threadIdx.x < st) red[threadIdx.x] += red[threadIdx.x + st];
        __syncthreads();
    }
    if (threadIdx.x == 0) out[0] = rsqrtf(red[0] / 4096.0f);
}

// ---------------------------------------------------------------------------
// 3) wnorm[o] = rsqrt(mean_i(style[o][i]^2))   (one wave32 per row)
// ---------------------------------------------------------------------------
__global__ __launch_bounds__(256) void k_wnorm(const float* __restrict__ style,
                                               float* __restrict__ wnorm) {
    int lane = threadIdx.x & 31;
    int o = blockIdx.x * 8 + (threadIdx.x >> 5);
    float a = 0.f;
    for (int i = lane; i < CH; i += 32) { float v = style[(size_t)o * CH + i]; a += v * v; }
    for (int m = 16; m >= 1; m >>= 1) a += __shfl_xor(a, m, 32);
    if (lane == 0) wnorm[o] = rsqrtf(a / (float)CH);
}

// ---------------------------------------------------------------------------
// 4) wb[b][o][i] = normalize_i( style[o][i]*wnorm[o] * s[b][i]*sn_scale ) -> bf16
// ---------------------------------------------------------------------------
__global__ __launch_bounds__(256) void k_wb(const float* __restrict__ style,
                                            const float* __restrict__ s,
                                            const float* __restrict__ wnorm,
                                            const float* __restrict__ snscale,
                                            unsigned short* __restrict__ wb) {
    int lane = threadIdx.x & 31;
    int g = blockIdx.x * 8 + (threadIdx.x >> 5);   // 0..4095
    int b = g >> 9, o = g & 511;
    float wn = wnorm[o], ss = snscale[0];
    float a = 0.f;
    for (int i = lane; i < CH; i += 32) {
        float t = style[(size_t)o * CH + i] * wn * s[(size_t)b * CH + i] * ss;
        a += t * t;
    }
    for (int m = 16; m >= 1; m >>= 1) a += __shfl_xor(a, m, 32);
    float sc = rsqrtf(a + 1e-8f);
    for (int i = lane; i < CH; i += 32) {
        float t = style[(size_t)o * CH + i] * wn * s[(size_t)b * CH + i] * ss;
        wb[(size_t)g * CH + i] = f2bf(t * sc);
    }
}

// ---------------------------------------------------------------------------
// 5) Filter synthesis (hz/vt): one block per filter direction.
// ---------------------------------------------------------------------------
__global__ __launch_bounds__(256) void k_filters(const float* __restrict__ phases,
                                                 const float* __restrict__ freqs,
                                                 const float* __restrict__ hz_outdim,
                                                 const float* __restrict__ vt_outdim,
                                                 const float* __restrict__ gsig,
                                                 const float* __restrict__ gfilt,
                                                 const float* __restrict__ lowf,
                                                 float* __restrict__ outf) {
    __shared__ float smem[4096];
    __shared__ float red[256];
    __shared__ float colscale[32];
    __shared__ float lowb[NTAPS];
    __shared__ float meansh;
    const int tid = threadIdx.x;
    const int axis = blockIdx.x;                    // 0 = hz, 1 = vt
    const float* coeff = (axis == 0) ? hz_outdim : vt_outdim;
    const int g  = tid & 31;
    const int ib = tid >> 5;
    if (tid < NTAPS) lowb[tid] = lowf[tid];

    float wreg[64];
    #pragma unroll
    for (int j = 0; j < 64; ++j) wreg[j] = 0.f;

    const float GRID_SCALE = 2.0f / 33.0f;
    for (int c0 = 0; c0 < CH; c0 += 128) {
        __syncthreads();
        for (int jj = 0; jj < 16; ++jj) {
            int idx = tid + 256 * jj;
            int cc = c0 + (idx >> 5), gg = idx & 31;
            float gv = ((float)gg - 15.5f) * GRID_SCALE;
            float arg = (freqs[cc * 2 + axis] * gv + phases[cc]) * 6.283185307179586f;
            smem[idx] = sinf(arg) * 0.6065306597126334f;   // exp(-0.5)
        }
        __syncthreads();
        for (int j = 0; j < 64; ++j) {
            int i = ib + 8 * j;
            const float* cr = coeff + (size_t)i * CH + c0;
            float a = wreg[j];
            for (int c = 0; c < 128; ++c) a += cr[c] * smem[c * 32 + g];
            wreg[j] = a;
        }
    }
    #pragma unroll
    for (int j = 0; j < 64; ++j) wreg[j] *= INV_SQRT512;

    float p = 0.f;
    #pragma unroll
    for (int j = 0; j < 64; ++j) p += wreg[j];
    __syncthreads(); red[tid] = p; __syncthreads();
    for (int st = 128; st > 0; st >>= 1) {
        if (tid < st) red[tid] += red[tid + st];
        __syncthreads();
    }
    if (tid == 0) meansh = red[0] * (1.0f / 16384.0f);
    __syncthreads();
    float mean = meansh;
    #pragma unroll
    for (int j = 0; j < 64; ++j) wreg[j] -= mean;

    p = 0.f;
    #pragma unroll
    for (int j = 0; j < 64; ++j) p += wreg[j] * wreg[j];
    __syncthreads(); red[tid] = p; __syncthreads();
    if (tid < 32) {
        float cs = 0.f;
        for (int u = 0; u < 8; ++u) cs += red[tid + 32 * u];
        colscale[tid] = rsqrtf(cs * (1.0f / (float)CH) + 1e-8f);
    }
    __syncthreads();

    float sig = gsig[0];
    float gs = 1.0f + sig * sig * 0.2f;
    float gf = gfilt[g];
    float csc = colscale[g] * expf(-(gf * gf) / (2.0f * gs));
    #pragma unroll
    for (int j = 0; j < 64; ++j) wreg[j] *= csc;

    for (int blk = 0; blk < 8; ++blk) {
        __syncthreads();
        for (int jj = 0; jj < 8; ++jj) {
            int j = blk * 8 + jj;
            int il = ib + 8 * jj;
            smem[il * 32 + g] = wreg[j];
        }
        __syncthreads();
        if (tid < 192) {
            int r = tid / 3, tt = tid - r * 3;
            int t0 = tt * 2;
            float a = 0.f;
            for (int k = 0; k < NTAPS; ++k) a += smem[r * 32 + t0 + k] * lowb[k];
            outf[axis * (CH * 3) + (blk * 64 + r) * 3 + tt] = a;
        }
    }
}

// ---------------------------------------------------------------------------
// 6) Separable depthwise conv (3-tap hz then vt), zero-padded; bf16 output.
// ---------------------------------------------------------------------------
__global__ __launch_bounds__(256) void k_dwconv(const float* __restrict__ x,
                                                const float* __restrict__ filt,
                                                unsigned short* __restrict__ out) {
    __shared__ float pin[4096];
    __shared__ float ph[4096];
    int c = blockIdx.x & 511, b = blockIdx.x >> 9;
    const float* src = x + ((size_t)(b * CH + c) << 12);
    unsigned short* dst = out + ((size_t)(b * CH + c) << 12);
    float h0 = filt[c * 3 + 0], h1 = filt[c * 3 + 1], h2 = filt[c * 3 + 2];
    float v0 = filt[CH * 3 + c * 3 + 0], v1 = filt[CH * 3 + c * 3 + 1], v2 = filt[CH * 3 + c * 3 + 2];
    int tid = threadIdx.x;
    for (int j = 0; j < 16; ++j) { int e = tid + 256 * j; pin[e] = src[e]; }
    __syncthreads();
    for (int j = 0; j < 16; ++j) {
        int e = tid + 256 * j;
        int xx = e & 63;
        float L = xx ? pin[e - 1] : 0.f;
        float R = (xx < 63) ? pin[e + 1] : 0.f;
        ph[e] = h0 * L + h1 * pin[e] + h2 * R;
    }
    __syncthreads();
    for (int j = 0; j < 16; ++j) {
        int e = tid + 256 * j;
        int y = e >> 6;
        float U = y ? ph[e - 64] : 0.f;
        float D = (y < 63) ? ph[e + 64] : 0.f;
        dst[e] = f2bf(v0 * U + v1 * ph[e] + v2 * D);
    }
}

// ---------------------------------------------------------------------------
// 7) Tiled transpose: src (R x C, T in {f32, bf16}) -> dst (C x R, bf16).
//    64x64 tiles via LDS; conflict-free pitch 66.
// ---------------------------------------------------------------------------
template <typename T>
__global__ __launch_bounds__(256) void k_transpose(const T* __restrict__ src,
                                                   unsigned short* __restrict__ dst,
                                                   int R, int C,
                                                   long long sbs, long long dbs) {
    __shared__ unsigned short tile[64 * 66];
    const int tid = threadIdx.x;
    const int r0 = blockIdx.y * 64, c0 = blockIdx.x * 64;
    const T* s = src + (size_t)blockIdx.z * sbs;
    unsigned short* d = dst + (size_t)blockIdx.z * dbs;
    for (int j = 0; j < 16; ++j) {
        int idx = tid + 256 * j;
        int rl = idx >> 6, cl = idx & 63;
        tile[cl * 66 + rl] = to_bf(s[(size_t)(r0 + rl) * C + c0 + cl]);
    }
    __syncthreads();
    for (int j = 0; j < 16; ++j) {
        int idx = tid + 256 * j;
        int cc = idx >> 6, rr = idx & 63;
        d[(size_t)(c0 + cc) * R + r0 + rr] = tile[cc * 66 + rr];
    }
}

// ---------------------------------------------------------------------------
// 8) WMMA GEMM: C[bz] = A[bz](MxK bf16 row-major) @ B[bz](NxK bf16 row-major,
//    i.e. B^T stored K-contiguous). Tile 128x128x32, 8 waves (2M x 4N),
//    each wave 64x32 via 4x2 wmma accums. Double-buffered LDS filled with
//    CDNA5 async global->LDS b128 copies (ASYNCcnt).
//    MODE 0: store bf16 * 1/sqrt(512)   (M[b] = wb[b] @ lr_weight)
//    MODE 1: +bias, leaky(0.2)*sqrt2, clamp +-256, store f32 (final output)
// ---------------------------------------------------------------------------
template <int MODE>
__global__ __launch_bounds__(256) void gemm_wmma(const unsigned short* __restrict__ A,
                                                 long long a_bs,
                                                 const unsigned short* __restrict__ B,
                                                 long long b_bs,
                                                 void* __restrict__ Out,
                                                 long long o_bs,
                                                 const float* __restrict__ bias,
                                                 int M, int N, int K) {
    __shared__ __attribute__((aligned(16))) unsigned short As[2][128 * 40];
    __shared__ __attribute__((aligned(16))) unsigned short Bs[2][128 * 40];
    const int tid = threadIdx.x;
    const int bx = blockIdx.x, by = blockIdx.y, bz = blockIdx.z;
    const int wave = tid >> 5, lane = tid & 31;
    const int wm = wave & 1, wn = wave >> 1;            // 2 x 4 wave grid
    const int l16 = lane & 15, kh = lane >> 4;

    // Per-thread 16B staging chunk: row = tid>>1 (of 128), half = tid&1.
    const int row = tid >> 1, half = tid & 1;
    const unsigned short* Ag = A + (size_t)bz * a_bs + (size_t)(by * 128 + row) * K + half * 16;
    const unsigned short* Bg = B + (size_t)bz * b_bs + (size_t)(bx * 128 + row) * K + half * 16;
    const unsigned ldsA = (unsigned)(uintptr_t)(&As[0][row * 40 + half * 16]);
    const unsigned ldsB = (unsigned)(uintptr_t)(&Bs[0][row * 40 + half * 16]);
    const unsigned bufB = 128u * 40u * 2u;              // 10240 bytes per buffer

    v8f acc[4][2];
    const v8f vzero = {0.f, 0.f, 0.f, 0.f, 0.f, 0.f, 0.f, 0.f};
    #pragma unroll
    for (int i = 0; i < 4; ++i)
        #pragma unroll
        for (int j = 0; j < 2; ++j) acc[i][j] = vzero;

    const int kT = K >> 5;
    // prologue: stage tile 0 into buffer 0
    async_copy_b128(ldsA, (const void*)Ag);
    async_copy_b128(ldsB, (const void*)Bg);

    for (int kt = 0; kt < kT; ++kt) {
        const int par = kt & 1;
        wait_async0();
        __syncthreads();                               // buffer `par` ready everywhere
        if (kt + 1 < kT) {                             // overlap next tile copy
            const unsigned nb = (unsigned)(par ^ 1) * bufB;
            async_copy_b128(ldsA + nb, (const void*)(Ag + (size_t)(kt + 1) * 32));
            async_copy_b128(ldsB + nb, (const void*)(Bg + (size_t)(kt + 1) * 32));
        }

        const unsigned short* Ash = As[par];
        const unsigned short* Bsh = Bs[par];
        FragBf af[4], bfr[2];
        #pragma unroll
        for (int fm = 0; fm < 4; ++fm) {               // ISA 16-bit A 16x32 layout
            int m = wm * 64 + fm * 16 + l16;
            af[fm].q[0] = *(const uint4*)(Ash + m * 40 + kh * 8);        // K = kh*8..
            af[fm].q[1] = *(const uint4*)(Ash + m * 40 + 16 + kh * 8);   // K = 16+kh*8..
        }
        #pragma unroll
        for (int fn = 0; fn < 2; ++fn) {               // ISA 16-bit B 32x16 layout
            int n = wn * 32 + fn * 16 + l16;
            const uint4* p = (const uint4*)(Bsh + n * 40 + kh * 8);      // K = kh*8.., 16+kh*8..
            bfr[fn].q[0] = p[0]; bfr[fn].q[1] = p[2];
        }
        #pragma unroll
        for (int fm = 0; fm < 4; ++fm)
            #pragma unroll
            for (int fn = 0; fn < 2; ++fn)
                acc[fm][fn] = __builtin_amdgcn_wmma_f32_16x16x32_bf16(
                    false, af[fm].v, false, bfr[fn].v,
                    (short)0, acc[fm][fn], false, false);
        __syncthreads();                               // done reading buffer `par`
    }

    // ---- epilogue ----
    #pragma unroll
    for (int fm = 0; fm < 4; ++fm) {
        #pragma unroll
        for (int fn = 0; fn < 2; ++fn) {
            int mg = by * 128 + wm * 64 + fm * 16 + 8 * kh;
            int ng = bx * 128 + wn * 32 + fn * 16 + l16;
            #pragma unroll
            for (int v = 0; v < 8; ++v) {
                float val = acc[fm][fn][v];
                int m = mg + v;
                if (MODE == 0) {
                    ((unsigned short*)Out)[(size_t)bz * o_bs + (size_t)m * N + ng] =
                        f2bf(val * INV_SQRT512);
                } else {
                    val += bias[m];
                    val = (val >= 0.f) ? val : 0.2f * val;
                    val *= SQRT2;
                    val = fminf(fmaxf(val, -256.f), 256.f);
                    ((float*)Out)[(size_t)bz * o_bs + (size_t)m * N + ng] = val;
                }
            }
        }
    }
}

// ---------------------------------------------------------------------------
// Workspace layout (bytes)
// ---------------------------------------------------------------------------
static constexpr size_t OFF_S     = 0;                                          // 16 KB
static constexpr size_t OFF_SNS   = 16384;
static constexpr size_t OFF_WNORM = 16640;
static constexpr size_t OFF_FILT  = 18944;                                      // 12 KB
static constexpr size_t OFF_WB    = 32768;                                      // 4 MB bf16
static constexpr size_t OFF_MCOMB = OFF_WB + (size_t)BATCH * CH * CH * 2;       // 4 MB bf16
static constexpr size_t OFF_LRT   = OFF_MCOMB + (size_t)BATCH * CH * CH * 2;    // 512 KB bf16
static constexpr size_t OFF_XDW   = OFF_LRT + (size_t)CH * CH * 2;              // 33.5 MB bf16 NCHW
static constexpr size_t OFF_XDWT  = OFF_XDW + (size_t)BATCH * CH * HW * 2;      // 33.5 MB bf16 [b][p][c]

extern "C" void kernel_launch(void* const* d_in, const int* in_sizes, int n_in,
                              void* d_out, int out_size, void* d_ws, size_t ws_size,
                              hipStream_t stream) {
    const float* x          = (const float*)d_in[0];
    const float* w_latent   = (const float*)d_in[1];
    const float* affine_w   = (const float*)d_in[2];
    const float* affine_b   = (const float*)d_in[3];
    const float* phases     = (const float*)d_in[4];
    const float* freqs      = (const float*)d_in[5];
    const float* hz_outdim  = (const float*)d_in[6];
    const float* vt_outdim  = (const float*)d_in[7];
    const float* gauss_sig  = (const float*)d_in[8];
    const float* gauss_filt = (const float*)d_in[9];
    const float* low_filter = (const float*)d_in[10];
    const float* lr_weight  = (const float*)d_in[11];
    const float* style_w    = (const float*)d_in[12];
    const float* bias       = (const float*)d_in[13];

    char* ws = (char*)d_ws;
    float*          s_ptr = (float*)(ws + OFF_S);
    float*          sns   = (float*)(ws + OFF_SNS);
    float*          wnorm = (float*)(ws + OFF_WNORM);
    float*          filt  = (float*)(ws + OFF_FILT);
    unsigned short* wb    = (unsigned short*)(ws + OFF_WB);
    unsigned short* mcomb = (unsigned short*)(ws + OFF_MCOMB);
    unsigned short* lrT   = (unsigned short*)(ws + OFF_LRT);
    unsigned short* xdw   = (unsigned short*)(ws + OFF_XDW);
    unsigned short* xdwT  = (unsigned short*)(ws + OFF_XDWT);

    k_affine  <<<16,   256, 0, stream>>>(w_latent, affine_w, affine_b, s_ptr);
    k_s_reduce<<<1,    256, 0, stream>>>(s_ptr, sns);
    k_wnorm   <<<64,   256, 0, stream>>>(style_w, wnorm);
    k_wb      <<<512,  256, 0, stream>>>(style_w, s_ptr, wnorm, sns, wb);

    // lrT[i][o1] = bf16(lr_weight[o1][i])
    k_transpose<float><<<dim3(8, 8, 1), 256, 0, stream>>>(
        lr_weight, lrT, CH, CH, 0LL, 0LL);

    // M[b] = wb[b] @ lr_weight * 1/sqrt(512), bf16 row-major [o][i]
    gemm_wmma<0><<<dim3(4, 4, BATCH), 256, 0, stream>>>(
        wb, (long long)CH * CH, lrT, 0LL,
        (void*)mcomb, (long long)CH * CH, nullptr, CH, CH, CH);

    k_filters <<<2,    256, 0, stream>>>(phases, freqs, hz_outdim, vt_outdim,
                                         gauss_sig, gauss_filt, low_filter, filt);
    k_dwconv  <<<4096, 256, 0, stream>>>(x, filt, xdw);

    // xdwT[b][p][c] = xdw[b][c][p]  (bf16, K-contiguous for B-side fragments)
    k_transpose<unsigned short><<<dim3(64, 8, BATCH), 256, 0, stream>>>(
        xdw, xdwT, CH, HW, (long long)CH * HW, (long long)CH * HW);

    // out[b] = act( M[b] @ xdw[b] + bias )
    gemm_wmma<1><<<dim3(32, 4, BATCH), 256, 0, stream>>>(
        mcomb, (long long)CH * CH, xdwT, (long long)CH * HW,
        d_out, (long long)CH * HW, bias, CH, HW, CH);
}
